// GPSLayer_32349693673617
// MI455X (gfx1250) — matrix operand order repaired and checked by reading, not compiled, not run on hardware
//
#include <hip/hip_runtime.h>
#include <hip/hip_bf16.h>

// ---------------------------------------------------------------------------
// GPS layer (GCN + MHA + FFN) for gfx1250 (MI455X).
// bf16 WMMA + fp32 accumulation; async global->LDS staging (ASYNCcnt).
// N=4096, D=256, H=4, HD=64, E=131072.
// ---------------------------------------------------------------------------

typedef __bf16 bf16;
typedef __attribute__((ext_vector_type(16))) __bf16 v16bf;
typedef __attribute__((ext_vector_type(8)))  __bf16 v8bf;
typedef __attribute__((ext_vector_type(4)))  __bf16 v4bf;
typedef __attribute__((ext_vector_type(8)))  float  v8f;

static __device__ inline v8f vzero8() {
  v8f z;
#pragma unroll
  for (int i = 0; i < 8; ++i) z[i] = 0.0f;
  return z;
}

static __device__ inline v8f wmma_bf16(v16bf a, v16bf b, v8f c) {
  return __builtin_amdgcn_wmma_f32_16x16x32_bf16(false, a, false, b, (short)0, c,
                                                 false, false);
}

// Generic pointer into LDS: low 32 bits are the DS byte offset (ISA 10.2).
static __device__ inline unsigned lds_off(const void* p) {
  return (unsigned)(size_t)p;
}

// Async copy 16B global -> LDS, tracked by ASYNCcnt.
static __device__ inline void async_copy_b128(unsigned dst_lds, const void* src) {
  asm volatile("global_load_async_to_lds_b128 %0, %1, off"
               :: "v"(dst_lds), "v"(src) : "memory");
}
static __device__ inline void wait_async0() {
  asm volatile("s_wait_asynccnt 0x0" ::: "memory");
}

// A fragment: 16x32 (MxK), row-major in LDS with stride ld (bf16 elems).
static __device__ inline v16bf load_frag_a(const bf16* base, int ld, int lane) {
  int m  = lane & 15;
  int hf = lane >> 4;
  const bf16* rowp = base + m * ld;
  v8bf lo = *(const v8bf*)(rowp + hf * 8);
  v8bf hi = *(const v8bf*)(rowp + 16 + hf * 8);
  v16bf a;
#pragma unroll
  for (int i = 0; i < 8; ++i) { a[i] = lo[i]; a[i + 8] = hi[i]; }
  return a;
}

// B fragment: 32x16 (KxN) stored transposed in LDS: Bt[n][k], stride ld.
static __device__ inline v16bf load_frag_b(const bf16* base, int ld, int lane) {
  int n  = lane & 15;
  int hf = lane >> 4;
  const bf16* rowp = base + n * ld + hf * 8;
  v8bf lo = *(const v8bf*)(rowp);
  v8bf hi = *(const v8bf*)(rowp + 16);
  v16bf b;
#pragma unroll
  for (int i = 0; i < 8; ++i) { b[i] = lo[i]; b[i + 8] = hi[i]; }
  return b;
}

// ---------------------------------------------------------------------------
// Weight convert + transpose: Wt[n*K + k] = (bf16)W[k*N + n]
// 2D grid (no integer division): blockIdx.x = k, blockIdx.y*256+tid = n.
// ---------------------------------------------------------------------------
__global__ __launch_bounds__(256) void transpose_f2b_kernel(
    const float* __restrict__ W, bf16* __restrict__ Wt, int K, int N)
{
  int k = blockIdx.x;
  int n = blockIdx.y * 256 + threadIdx.x;
  if (n < N) Wt[(size_t)n * K + k] = (bf16)W[(size_t)k * N + n];
}

__global__ __launch_bounds__(256) void f2b_kernel(const float* __restrict__ in,
                                                  bf16* __restrict__ out, long n) {
  long i = (long)blockIdx.x * blockDim.x + threadIdx.x;
  if (i < n) out[i] = (bf16)in[i];
}

// ---------------------------------------------------------------------------
// GEMM: C[M,N] = act(A[M,K] @ B[K,N] + bias), A bf16 row-major,
// B given pre-transposed bf16 Bt[N,K]. Staging via async global->LDS b128.
// Block tile 128x64, K-step 64, 8 waves (4Mx2N), 8 WMMAs per K-step.
// Output fp32 (Cf) or bf16 (Cb) -- exactly one non-null.
// Requires M%128==0, N%64==0, K%64==0.
// ---------------------------------------------------------------------------
#define GTM 128
#define GTN 64
#define GTK 64
#define LDA_S 72   // 64 + 8 pad; 144B row stride (16B aligned)
#define LDB_S 72

__global__ __launch_bounds__(256) void gemm_bf16_kernel(
    const bf16* __restrict__ A, const bf16* __restrict__ Bt,
    const float* __restrict__ bias, float* __restrict__ Cf,
    bf16* __restrict__ Cb, int M, int N, int K, int relu)
{
  __shared__ __align__(16) bf16 As[GTM * LDA_S];
  __shared__ __align__(16) bf16 Bs[GTN * LDB_S];

  const int tid   = threadIdx.x;
  const int lane  = tid & 31;
  const int wave  = tid >> 5;      // 0..7
  const int waveM = wave & 3;      // 4 waves along M
  const int waveN = wave >> 2;     // 2 waves along N
  const int bm = blockIdx.x * GTM;
  const int bn = blockIdx.y * GTN;

  v8f acc[2][2];
#pragma unroll
  for (int mt = 0; mt < 2; ++mt)
#pragma unroll
    for (int nt = 0; nt < 2; ++nt) acc[mt][nt] = vzero8();

  for (int k0 = 0; k0 < K; k0 += GTK) {
    if (k0 + GTK < K) {
      __builtin_prefetch(&A[(size_t)(bm + (tid & 127)) * K + k0 + GTK], 0, 1);
      __builtin_prefetch(&Bt[(size_t)(bn + (tid & 63)) * K + k0 + GTK], 0, 1);
    }
    // Async stage A tile: 128 rows x 64 bf16 = 8 chunks of 16B per row.
    {
      int c  = tid & 7;             // chunk in row
      int r0 = tid >> 3;            // 0..31
#pragma unroll
      for (int i = 0; i < 4; ++i) {
        int r = r0 + i * 32;
        async_copy_b128(lds_off(&As[r * LDA_S + c * 8]),
                        &A[(size_t)(bm + r) * K + k0 + c * 8]);
      }
    }
    // Async stage Bt tile: 64 rows (n) x 64 bf16 (k).
    {
      int c  = tid & 7;
      int n0 = tid >> 3;            // 0..31
#pragma unroll
      for (int i = 0; i < 2; ++i) {
        int n = n0 + i * 32;
        async_copy_b128(lds_off(&Bs[n * LDB_S + c * 8]),
                        &Bt[(size_t)(bn + n) * K + k0 + c * 8]);
      }
    }
    wait_async0();
    __syncthreads();

#pragma unroll
    for (int ks = 0; ks < 2; ++ks) {
      v16bf af[2], bf[2];
#pragma unroll
      for (int mt = 0; mt < 2; ++mt)
        af[mt] = load_frag_a(&As[(waveM * 32 + mt * 16) * LDA_S + ks * 32],
                             LDA_S, lane);
#pragma unroll
      for (int nt = 0; nt < 2; ++nt)
        bf[nt] = load_frag_b(&Bs[(waveN * 32 + nt * 16) * LDB_S + ks * 32],
                             LDB_S, lane);
#pragma unroll
      for (int mt = 0; mt < 2; ++mt)
#pragma unroll
        for (int nt = 0; nt < 2; ++nt)
          acc[mt][nt] = wmma_bf16(af[mt], bf[nt], acc[mt][nt]);
    }
    __syncthreads();
  }

  // Epilogue: lane L, vgpr r: m = r + (L/16)*8, n = L%16
  const int hf  = lane >> 4;
  const int nlo = lane & 15;
#pragma unroll
  for (int mt = 0; mt < 2; ++mt) {
#pragma unroll
    for (int nt = 0; nt < 2; ++nt) {
      int col = bn + waveN * 32 + nt * 16 + nlo;
      float bv = bias ? bias[col] : 0.0f;
#pragma unroll
      for (int r = 0; r < 8; ++r) {
        int row = bm + waveM * 32 + mt * 16 + r + hf * 8;
        float v = acc[mt][nt][r] + bv;
        if (relu) v = fmaxf(v, 0.0f);
        if (Cb) Cb[(size_t)row * N + col] = (bf16)v;
        else    Cf[(size_t)row * N + col] = v;
      }
    }
  }
}

// ---------------------------------------------------------------------------
// Degree / scatter kernels for GCN (fp32 domain)
// ---------------------------------------------------------------------------
__global__ __launch_bounds__(256) void deg_init_kernel(float* deg, int n) {
  int i = blockIdx.x * blockDim.x + threadIdx.x;
  if (i < n) deg[i] = 1.0f;   // self loop
}

__global__ __launch_bounds__(256) void deg_count_kernel(const int* __restrict__ dst,
                                                        float* deg, int e) {
  int i = blockIdx.x * blockDim.x + threadIdx.x;
  if (i < e) atomicAdd(&deg[dst[i]], 1.0f);
}

__global__ __launch_bounds__(256) void deg_rsqrt_kernel(float* deg, int n) {
  int i = blockIdx.x * blockDim.x + threadIdx.x;
  if (i < n) deg[i] = rsqrtf(deg[i]);   // in place -> dinv
}

__global__ __launch_bounds__(256) void gcn_scatter_kernel(
    const float* __restrict__ h, const int* __restrict__ src,
    const int* __restrict__ dst, const float* __restrict__ dinv,
    float* xl, long total)
{
  long idx = (long)blockIdx.x * blockDim.x + threadIdx.x;
  if (idx >= total) return;
  int e = (int)(idx >> 8);
  int c = (int)(idx & 255);
  int s = src[e], d = dst[e];
  float w = dinv[s] * dinv[d];
  atomicAdd(&xl[(size_t)d * 256 + c], h[(size_t)s * 256 + c] * w);
}

// ---------------------------------------------------------------------------
// GCN finish (+self loop +bias) + residual + LN; dual fp32/bf16 output.
// ---------------------------------------------------------------------------
__global__ __launch_bounds__(256) void gcn_ln_kernel(
    const float* __restrict__ x, const float* __restrict__ xl,
    const float* __restrict__ h, const float* __restrict__ dinv,
    const float* __restrict__ b, const float* __restrict__ g,
    const float* __restrict__ beta, float* __restrict__ out,
    bf16* __restrict__ outb)
{
  const int row = blockIdx.x, c = threadIdx.x, D = 256;
  float di = dinv[row];
  float t = x[(size_t)row * D + c] + xl[(size_t)row * D + c]
          + h[(size_t)row * D + c] * di * di + b[c];
  __shared__ float s1[256], s2[256];
  s1[c] = t; s2[c] = t * t;
  __syncthreads();
  for (int s = 128; s > 0; s >>= 1) {
    if (c < s) { s1[c] += s1[c + s]; s2[c] += s2[c + s]; }
    __syncthreads();
  }
  float mu  = s1[0] * (1.0f / 256.0f);
  float var = s2[0] * (1.0f / 256.0f) - mu * mu;
  float v = (t - mu) * rsqrtf(var + 1e-5f) * g[c] + beta[c];
  out[(size_t)row * D + c] = v;
  if (outb) outb[(size_t)row * D + c] = (bf16)v;
}

__global__ __launch_bounds__(256) void add_ln_kernel(
    const float* __restrict__ a, const float* __restrict__ bsrc,
    const float* __restrict__ g, const float* __restrict__ beta,
    float* __restrict__ out, bf16* __restrict__ outb)
{
  const int row = blockIdx.x, c = threadIdx.x, D = 256;
  float t = a[(size_t)row * D + c] + bsrc[(size_t)row * D + c];
  __shared__ float s1[256], s2[256];
  s1[c] = t; s2[c] = t * t;
  __syncthreads();
  for (int s = 128; s > 0; s >>= 1) {
    if (c < s) { s1[c] += s1[c + s]; s2[c] += s2[c + s]; }
    __syncthreads();
  }
  float mu  = s1[0] * (1.0f / 256.0f);
  float var = s2[0] * (1.0f / 256.0f) - mu * mu;
  float v = (t - mu) * rsqrtf(var + 1e-5f) * g[c] + beta[c];
  out[(size_t)row * D + c] = v;
  if (outb) outb[(size_t)row * D + c] = (bf16)v;
}

// ---------------------------------------------------------------------------
// Flash attention over bf16 qkv [N, 768]; one (head, 64 Q rows) per block.
// Q/K tiles staged via async global->LDS b128; V staged transposed (bf16 copy).
// Output written as bf16 (consumed only by the out-projection GEMM).
// ---------------------------------------------------------------------------
#define AQ 64
#define AK 64
#define ALD 72   // 64 + 8 pad, 144B stride (16B aligned)

__global__ __launch_bounds__(128) void attn_kernel(
    const bf16* __restrict__ qkv, bf16* __restrict__ ob, int N)
{
  __shared__ __align__(16) bf16 Qs[AQ * ALD];
  __shared__ __align__(16) bf16 Ks[AK * ALD];
  __shared__ __align__(16) bf16 Vts[64 * ALD];   // Vt[dim][key]
  __shared__ __align__(16) bf16 Ps[AQ * ALD];
  __shared__ float mrun[AQ], lrun[AQ];

  const int tid  = threadIdx.x;
  const int lane = tid & 31;
  const int wave = tid >> 5;          // 0..3
  const int hd   = blockIdx.x & 3;    // head (H=4)
  const int qb   = (blockIdx.x >> 2) * AQ;
  const int D3   = 768;
  const int hf   = lane >> 4;
  const int nlo  = lane & 15;
  const float scale = 0.125f;         // 1/sqrt(64)

  // Async stage Q block: 64 rows x 64 bf16 = 8 chunks of 16B per row.
  {
    int c  = tid & 7;
    int r0 = tid >> 3;                // 0..15
#pragma unroll
    for (int i = 0; i < 4; ++i) {
      int r = r0 + i * 16;
      async_copy_b128(lds_off(&Qs[r * ALD + c * 8]),
                      &qkv[(size_t)(qb + r) * D3 + hd * 64 + c * 8]);
    }
  }
  if (tid < AQ) { mrun[tid] = -1e30f; lrun[tid] = 0.0f; }

  v8f oacc[4];
#pragma unroll
  for (int nt = 0; nt < 4; ++nt) oacc[nt] = vzero8();
  __syncthreads();

  for (int kt = 0; kt < N; kt += AK) {
    if (kt + AK < N)
      __builtin_prefetch(&qkv[(size_t)(kt + AK + (tid >> 1)) * D3 + 256 + hd * 64],
                         0, 1);
    // Async stage K tile (row-major = Bt layout for Q@K^T)
    {
      int c  = tid & 7;
      int r0 = tid >> 3;
#pragma unroll
      for (int i = 0; i < 4; ++i) {
        int r = r0 + i * 16;
        async_copy_b128(lds_off(&Ks[r * ALD + c * 8]),
                        &qkv[(size_t)(kt + r) * D3 + 256 + hd * 64 + c * 8]);
      }
    }
    // Stage V transposed: Vt[dim][key]; 4 consecutive keys per b64 store.
    {
      int d   = tid & 63;
      int rg0 = tid >> 6;             // 0..1
#pragma unroll
      for (int i = 0; i < 8; ++i) {
        int rg = rg0 + i * 2;         // 0..15 (key groups of 4)
        v4bf pk;
        pk[0] = qkv[(size_t)(kt + rg * 4 + 0) * D3 + 512 + hd * 64 + d];
        pk[1] = qkv[(size_t)(kt + rg * 4 + 1) * D3 + 512 + hd * 64 + d];
        pk[2] = qkv[(size_t)(kt + rg * 4 + 2) * D3 + 512 + hd * 64 + d];
        pk[3] = qkv[(size_t)(kt + rg * 4 + 3) * D3 + 512 + hd * 64 + d];
        *(v4bf*)&Vts[d * ALD + rg * 4] = pk;
      }
    }
    wait_async0();
    __syncthreads();

    // S = Q @ K^T  (16x64 per wave, 4 n-tiles, K-dim 64 = 2 WMMA steps)
    v8f s[4];
#pragma unroll
    for (int nt = 0; nt < 4; ++nt) s[nt] = vzero8();
#pragma unroll
    for (int ks = 0; ks < 2; ++ks) {
      v16bf a = load_frag_a(&Qs[(wave * 16) * ALD + ks * 32], ALD, lane);
#pragma unroll
      for (int nt = 0; nt < 4; ++nt) {
        v16bf b = load_frag_b(&Ks[(nt * 16) * ALD + ks * 32], ALD, lane);
        s[nt] = wmma_bf16(a, b, s[nt]);
      }
    }

    // Online softmax per row; row m = wave*16 + hf*8 + r in one 16-lane half
    float ralpha[8];
#pragma unroll
    for (int r = 0; r < 8; ++r) {
      int m = wave * 16 + hf * 8 + r;
      float mx = -1e30f;
#pragma unroll
      for (int nt = 0; nt < 4; ++nt) {
        float v = s[nt][r] * scale;
        s[nt][r] = v;
        mx = fmaxf(mx, v);
      }
#pragma unroll
      for (int off = 1; off < 16; off <<= 1)
        mx = fmaxf(mx, __shfl_xor(mx, off, 32));
      float mold = mrun[m];
      float mnew = fmaxf(mold, mx);
      float al   = __expf(mold - mnew);
      float rsum = 0.0f;
#pragma unroll
      for (int nt = 0; nt < 4; ++nt) {
        float p = __expf(s[nt][r] - mnew);
        s[nt][r] = p;
        rsum += p;
      }
#pragma unroll
      for (int off = 1; off < 16; off <<= 1)
        rsum += __shfl_xor(rsum, off, 32);
      if (nlo == 0) { mrun[m] = mnew; lrun[m] = al * lrun[m] + rsum; }
      ralpha[r] = al;
#pragma unroll
      for (int nt = 0; nt < 4; ++nt)
        Ps[m * ALD + nt * 16 + nlo] = (bf16)s[nt][r];
    }

    // Rescale running O by alpha
#pragma unroll
    for (int nt = 0; nt < 4; ++nt)
#pragma unroll
      for (int r = 0; r < 8; ++r) oacc[nt][r] *= ralpha[r];

    __syncthreads();  // Ps visible before PV

    // O += P @ V   (A = Ps rows of this wave, B = Vt[dim][key])
#pragma unroll
    for (int ks = 0; ks < 2; ++ks) {
      v16bf a = load_frag_a(&Ps[(wave * 16) * ALD + ks * 32], ALD, lane);
#pragma unroll
      for (int nt = 0; nt < 4; ++nt) {
        v16bf b = load_frag_b(&Vts[(nt * 16) * ALD + ks * 32], ALD, lane);
        oacc[nt] = wmma_bf16(a, b, oacc[nt]);
      }
    }
    __syncthreads();  // before next tile overwrites Ks/Vts
  }

  // Finalize: divide by row sum, write bf16 [row, hd*64 + dim]
#pragma unroll
  for (int r = 0; r < 8; ++r) {
    int m = wave * 16 + hf * 8 + r;
    float linv = 1.0f / lrun[m];
#pragma unroll
    for (int nt = 0; nt < 4; ++nt) {
      int row = qb + m;
      int col = hd * 64 + nt * 16 + nlo;
      ob[(size_t)row * 256 + col] = (bf16)(oacc[nt][r] * linv);
    }
  }
}

// ---------------------------------------------------------------------------
// Host launch
// ---------------------------------------------------------------------------
extern "C" void kernel_launch(void* const* d_in, const int* in_sizes, int n_in,
                              void* d_out, int out_size, void* d_ws, size_t ws_size,
                              hipStream_t stream) {
  const int N = 4096, D = 256;
  const int E = in_sizes[1] / 2;

  const float* x      = (const float*)d_in[0];
  const int*   edge   = (const int*)d_in[1];
  const float* Wgcn   = (const float*)d_in[2];
  const float* bgcn   = (const float*)d_in[3];
  const float* wqkv   = (const float*)d_in[4];
  const float* bqkv   = (const float*)d_in[5];
  const float* wout   = (const float*)d_in[6];
  const float* bout   = (const float*)d_in[7];
  const float* g1l    = (const float*)d_in[8];
  const float* beta1l = (const float*)d_in[9];
  const float* g1a    = (const float*)d_in[10];
  const float* beta1a = (const float*)d_in[11];
  const float* W1     = (const float*)d_in[12];
  const float* bf1    = (const float*)d_in[13];
  const float* W2     = (const float*)d_in[14];
  const float* bf2    = (const float*)d_in[15];
  const float* g2     = (const float*)d_in[16];
  const float* beta2  = (const float*)d_in[17];
  float* out = (float*)d_out;

  // Workspace carve-up (256B aligned regions)
  char* wsp = (char*)d_ws;
  auto alloc = [&](size_t bytes) {
    char* r = wsp;
    wsp += (bytes + 255) & ~(size_t)255;
    return r;
  };
  float* dinv   = (float*)alloc((size_t)N * 4);
  bf16*  xb     = (bf16*) alloc((size_t)N * D * 2);
  float* hbuf   = (float*)alloc((size_t)N * D * 4);
  float* xl     = (float*)alloc((size_t)N * D * 4);
  float* x1     = (float*)alloc((size_t)N * D * 4);
  bf16*  x1b    = (bf16*) alloc((size_t)N * D * 2);
  bf16*  qkvb   = (bf16*) alloc((size_t)N * 3 * D * 2);
  bf16*  xattnb = (bf16*) alloc((size_t)N * D * 2);
  float* aproj  = (float*)alloc((size_t)N * D * 4);
  float* x2     = (float*)alloc((size_t)N * D * 4);
  bf16*  x2b    = (bf16*) alloc((size_t)N * D * 2);
  bf16*  f1b    = (bf16*) alloc((size_t)N * 2 * D * 2);
  float* f2     = (float*)alloc((size_t)N * D * 4);
  bf16*  wgT    = (bf16*) alloc((size_t)D * D * 2);        // [D][D]
  bf16*  wqkvT  = (bf16*) alloc((size_t)3 * D * D * 2);    // [3D][D]
  bf16*  woutT  = (bf16*) alloc((size_t)D * D * 2);        // [D][D]
  bf16*  w1T    = (bf16*) alloc((size_t)2 * D * D * 2);    // [2D][D]
  bf16*  w2T    = (bf16*) alloc((size_t)D * 2 * D * 2);    // [D][2D]

  const int* esrc = edge;
  const int* edst = edge + E;

  hipMemsetAsync(xl, 0, (size_t)N * D * sizeof(float), stream);

  // Weight convert + pre-transpose (bf16, [N][K] layout for async staging)
  transpose_f2b_kernel<<<dim3(D, D / 256), 256, 0, stream>>>(Wgcn, wgT, D, D);
  transpose_f2b_kernel<<<dim3(D, (3 * D) / 256), 256, 0, stream>>>(wqkv, wqkvT, D, 3 * D);
  transpose_f2b_kernel<<<dim3(D, D / 256), 256, 0, stream>>>(wout, woutT, D, D);
  transpose_f2b_kernel<<<dim3(D, (2 * D) / 256), 256, 0, stream>>>(W1, w1T, D, 2 * D);
  transpose_f2b_kernel<<<dim3(2 * D, D / 256), 256, 0, stream>>>(W2, w2T, 2 * D, D);
  f2b_kernel<<<(N * D + 255) / 256, 256, 0, stream>>>(x, xb, (long)N * D);

  // --- GCN ---
  deg_init_kernel<<<(N + 255) / 256, 256, 0, stream>>>(dinv, N);
  deg_count_kernel<<<(E + 255) / 256, 256, 0, stream>>>(edst, dinv, E);
  deg_rsqrt_kernel<<<(N + 255) / 256, 256, 0, stream>>>(dinv, N);

  gemm_bf16_kernel<<<dim3(N / GTM, D / GTN), 256, 0, stream>>>(
      xb, wgT, nullptr, hbuf, nullptr, N, D, D, 0);

  long total = (long)E * D;
  gcn_scatter_kernel<<<(unsigned)((total + 255) / 256), 256, 0, stream>>>(
      hbuf, esrc, edst, dinv, xl, total);

  gcn_ln_kernel<<<N, 256, 0, stream>>>(x, xl, hbuf, dinv, bgcn, g1l, beta1l,
                                       x1, x1b);

  // --- Attention ---
  gemm_bf16_kernel<<<dim3(N / GTM, (3 * D) / GTN), 256, 0, stream>>>(
      x1b, wqkvT, bqkv, nullptr, qkvb, N, 3 * D, D, 0);

  attn_kernel<<<4 * (N / AQ), 128, 0, stream>>>(qkvb, xattnb, N);

  gemm_bf16_kernel<<<dim3(N / GTM, D / GTN), 256, 0, stream>>>(
      xattnb, woutT, bout, aproj, nullptr, N, D, D, 0);

  add_ln_kernel<<<N, 256, 0, stream>>>(x1, aproj, g1a, beta1a, x2, x2b);

  // --- FFN ---
  gemm_bf16_kernel<<<dim3(N / GTM, (2 * D) / GTN), 256, 0, stream>>>(
      x2b, w1T, bf1, nullptr, f1b, N, 2 * D, D, 1);
  gemm_bf16_kernel<<<dim3(N / GTM, D / GTN), 256, 0, stream>>>(
      f1b, w2T, bf2, f2, nullptr, N, D, 2 * D, 0);

  add_ln_kernel<<<N, 256, 0, stream>>>(x2, f2, g2, beta2, out, nullptr);
}
